// Encoder_eCOLGCN_42356967473565
// MI455X (gfx1250) — matrix operand order repaired and checked by reading, not compiled, not run on hardware
//
#include <hip/hip_runtime.h>
#include <hip/hip_bf16.h>

// ---------------------------------------------------------------------------
// GCN encoder for MI455X (gfx1250, wave32, WMMA).
//   h1 = tanh( Â x W1 + b1 ),  h2 = tanh( Â h1 W2 + b2 ),  logits = h2 W3 + b3
//   Â = D^-1/2 (A + I) D^-1/2
// GEMMs via v_wmma_f32_16x16x32_f16 with f16 hi/lo split (3 WMMAs ~= fp32).
// Scatter phase: wave-per-edge, float4 gather, global_atomic_add_f32.
// ---------------------------------------------------------------------------

typedef __attribute__((ext_vector_type(16))) _Float16 v16h;
typedef __attribute__((ext_vector_type(8)))  float    v8f;

#define DIMH 128   // hidden / output feature width (both layers)

// ---------------- degree / normalization ----------------

__global__ void init_deg_kernel(float* __restrict__ deg, int n) {
    int i = blockIdx.x * blockDim.x + threadIdx.x;
    if (i < n) deg[i] = 1.0f;                       // self-loop contribution
}

__global__ void deg_edges_kernel(const int* __restrict__ dst, float* __restrict__ deg, int nE) {
    int e = blockIdx.x * blockDim.x + threadIdx.x;
    if (e < nE) atomicAdd(&deg[dst[e]], 1.0f);
}

__global__ void dis_kernel(const float* __restrict__ deg, float* __restrict__ dis, int n) {
    int i = blockIdx.x * blockDim.x + threadIdx.x;
    if (i < n) {
        float d = deg[i];
        dis[i] = (d > 0.0f) ? rsqrtf(d) : 0.0f;     // d >= 1 always (self-loop)
    }
}

// ---------------- WMMA GEMM:  H[n x 128] = X[n x K] @ W[K x 128] ------------
// Block = 256 threads = 8 waves. Block covers 16 rows; wave w covers cols
// [16w, 16w+16). f32 inputs split into f16 hi/lo; acc in f32 via 3 WMMAs/step.

template <int K>
__global__ __launch_bounds__(256)
void gemm_wmma_kernel(const float* __restrict__ X, const float* __restrict__ W,
                      float* __restrict__ H, int nRows) {
    __shared__ float As[16][36];     // 16 x 32 k-slab of X (padded: 33+ avoids conflicts)
    __shared__ float Bs[32][132];    // 32 x 128 k-slab of W (padded)

    const int tid  = threadIdx.x;
    const int wave = tid >> 5;
    const int lane = tid & 31;
    const int m0   = blockIdx.x * 16;
    const int n0   = wave * 16;
    const int mrow = lane & 15;          // A row / B,C column sub-index
    const int hiL  = (lane >> 4) & 1;    // lane group 0 or 1

    v8f acc = {};

    for (int k0 = 0; k0 < K; k0 += 32) {
        // ---- stage A tile: 512 floats, 2 per thread
        {
            int i0 = tid * 2;
            int r = i0 >> 5, c = i0 & 31;
            int row = m0 + r; if (row >= nRows) row = nRows - 1;   // uniform-safe clamp
            const float* xp = X + (size_t)row * K + k0 + c;
            As[r][c]     = xp[0];
            As[r][c + 1] = xp[1];
        }
        // ---- stage B tile: 4096 floats, 16 per thread
        #pragma unroll
        for (int i = 0; i < 16; ++i) {
            int idx = tid + i * 256;
            int r = idx >> 7, c = idx & 127;
            Bs[r][c] = W[(size_t)(k0 + r) * DIMH + c];
        }
        __syncthreads();

        // ---- build f16 hi/lo fragments (ISA 16-bit A/B wave32 layouts)
        v16h ahi, alo, bhi, blo;
        #pragma unroll
        for (int e = 0; e < 16; ++e) {
            // A 16x32: lanes<16 hold K {0..7,16..23}; lanes>=16 hold {8..15,24..31}
            int kkA = (e < 8 ? e : 8 + e) + (hiL ? 8 : 0);
            float a = As[mrow][kkA];
            _Float16 ah = (_Float16)a;
            ahi[e] = ah;
            alo[e] = (_Float16)(a - (float)ah);
            // B 32x16: lanes<16 hold K {0..7,16..23}; lanes>=16 hold {8..15,24..31} (N = lane&15)
            int kkB = (e < 8 ? e : 8 + e) + (hiL ? 8 : 0);
            float b = Bs[kkB][n0 + mrow];
            _Float16 bh = (_Float16)b;
            bhi[e] = bh;
            blo[e] = (_Float16)(b - (float)bh);
        }

        acc = __builtin_amdgcn_wmma_f32_16x16x32_f16(false, ahi, false, bhi,
                                                     (short)0, acc, false, false);
        acc = __builtin_amdgcn_wmma_f32_16x16x32_f16(false, ahi, false, blo,
                                                     (short)0, acc, false, false);
        acc = __builtin_amdgcn_wmma_f32_16x16x32_f16(false, alo, false, bhi,
                                                     (short)0, acc, false, false);
        __syncthreads();
    }

    // ---- store C: VGPR r -> (M = r + 8*hiL, N = lane&15)
    #pragma unroll
    for (int r = 0; r < 8; ++r) {
        int row = m0 + r + hiL * 8;
        int col = n0 + mrow;
        if (row < nRows) H[(size_t)row * DIMH + col] = acc[r];
    }
}

// ---------------- aggregation ----------------

// out[n][d] = H[n][d] * dis[n]^2  (self-loop)  + b[d]
__global__ void agg_init_kernel(const float* __restrict__ H, const float* __restrict__ dis,
                                const float* __restrict__ b, float* __restrict__ out, int n) {
    int idx = blockIdx.x * blockDim.x + threadIdx.x;
    if (idx >= n * DIMH) return;
    int row = idx >> 7, d = idx & (DIMH - 1);
    float s = dis[row];
    out[idx] = H[idx] * s * s + b[d];
}

// one wave per edge: gather float4 of h[src], scale, atomic-add into out[dst]
__global__ __launch_bounds__(256)
void scatter_edges_kernel(const float* __restrict__ H, const float* __restrict__ dis,
                          const int* __restrict__ src, const int* __restrict__ dst,
                          float* __restrict__ out, int nE) {
    int e = blockIdx.x * (blockDim.x >> 5) + (threadIdx.x >> 5);
    if (e >= nE) return;
    int lane = threadIdx.x & 31;
    int s = src[e], d = dst[e];
    float w = dis[s] * dis[d];
    const float4 v = ((const float4*)(H + (size_t)s * DIMH))[lane];
    float* od = out + (size_t)d * DIMH + lane * 4;
    atomicAdd(od + 0, v.x * w);
    atomicAdd(od + 1, v.y * w);
    atomicAdd(od + 2, v.z * w);
    atomicAdd(od + 3, v.w * w);
}

__global__ void tanh_kernel(float* __restrict__ p, int count) {
    int idx = blockIdx.x * blockDim.x + threadIdx.x;
    if (idx < count) p[idx] = tanhf(p[idx]);
}

// ---------------- logits: [n x 128] @ [128 x 8] + b3 ------------------------
__global__ void logits_kernel(const float* __restrict__ h2, const float* __restrict__ W3,
                              const float* __restrict__ b3, float* __restrict__ out, int n) {
    int idx = blockIdx.x * blockDim.x + threadIdx.x;
    if (idx >= n * 8) return;
    int row = idx >> 3, t = idx & 7;
    const float4* hp = (const float4*)(h2 + (size_t)row * DIMH);
    float acc = b3[t];
    #pragma unroll
    for (int i = 0; i < 32; ++i) {
        float4 v = hp[i];
        acc += v.x * W3[(i * 4 + 0) * 8 + t];
        acc += v.y * W3[(i * 4 + 1) * 8 + t];
        acc += v.z * W3[(i * 4 + 2) * 8 + t];
        acc += v.w * W3[(i * 4 + 3) * 8 + t];
    }
    out[idx] = acc;
}

// ---------------------------------------------------------------------------

extern "C" void kernel_launch(void* const* d_in, const int* in_sizes, int n_in,
                              void* d_out, int out_size, void* d_ws, size_t ws_size,
                              hipStream_t stream) {
    const float* x   = (const float*)d_in[0];
    const int*   ei  = (const int*)  d_in[1];
    const float* W1  = (const float*)d_in[2];
    const float* b1  = (const float*)d_in[3];
    const float* W2  = (const float*)d_in[4];
    const float* b2  = (const float*)d_in[5];
    const float* W3  = (const float*)d_in[6];
    const float* b3  = (const float*)d_in[7];

    const int IN_DIM = 256;
    const int n  = in_sizes[0] / IN_DIM;   // 100000
    const int nE = in_sizes[1] / 2;        // 1600000
    const int* src = ei;
    const int* dst = ei + nE;

    // workspace layout (~103.2 MB): deg | dis | bufA | bufB
    char* ws = (char*)d_ws;
    size_t off = 0;
    float* deg  = (float*)(ws + off); off += (size_t)n * sizeof(float);
    off = (off + 255) & ~(size_t)255;
    float* dis  = (float*)(ws + off); off += (size_t)n * sizeof(float);
    off = (off + 255) & ~(size_t)255;
    float* bufA = (float*)(ws + off); off += (size_t)n * DIMH * sizeof(float);
    off = (off + 255) & ~(size_t)255;
    float* bufB = (float*)(ws + off);

    float* h2     = (float*)d_out;              // [n x 128]
    float* logits = h2 + (size_t)n * DIMH;      // [n x 8]

    const int nF = n * DIMH;
    const int tileBlocks  = (n + 15) / 16;
    const int edgeBlocks  = (nE + 7) / 8;       // 8 waves per block

    // normalization
    init_deg_kernel <<<(n  + 255) / 256, 256, 0, stream>>>(deg, n);
    deg_edges_kernel<<<(nE + 255) / 256, 256, 0, stream>>>(dst, deg, nE);
    dis_kernel      <<<(n  + 255) / 256, 256, 0, stream>>>(deg, dis, n);

    // layer 1
    gemm_wmma_kernel<256><<<tileBlocks, 256, 0, stream>>>(x, W1, bufA, n);
    agg_init_kernel      <<<(nF + 255) / 256, 256, 0, stream>>>(bufA, dis, b1, bufB, n);
    scatter_edges_kernel <<<edgeBlocks, 256, 0, stream>>>(bufA, dis, src, dst, bufB, nE);
    tanh_kernel          <<<(nF + 255) / 256, 256, 0, stream>>>(bufB, nF);

    // layer 2 (aggregate directly into d_out h2 region)
    gemm_wmma_kernel<128><<<tileBlocks, 256, 0, stream>>>(bufB, W2, bufA, n);
    agg_init_kernel      <<<(nF + 255) / 256, 256, 0, stream>>>(bufA, dis, b2, h2, n);
    scatter_edges_kernel <<<edgeBlocks, 256, 0, stream>>>(bufA, dis, src, dst, h2, nE);
    tanh_kernel          <<<(nF + 255) / 256, 256, 0, stream>>>(h2, nF);

    // classifier head
    logits_kernel<<<(n * 8 + 255) / 256, 256, 0, stream>>>(h2, W3, b3, logits, n);
}
